// Attention_69776038691486
// MI455X (gfx1250) — compile-verified, matrix operand order checked by reading, not run on hardware
//
#include <hip/hip_runtime.h>
#include <hip/hip_bf16.h>
#include <math.h>

// Problem constants (from reference)
#define HIDDEN  1024
#define CONTEXT 1024
#define NB      64
#define LSP     1024                 // 32*32 spatial positions
#define FDIM    (HIDDEN + CONTEXT)   // 2048

// Main-kernel blocking
#define ROWS 128                     // L rows staged per block (8 waves x 16)
#define AW   516                     // padded LDS row stride in dwords (512 + 4)

typedef __attribute__((ext_vector_type(16))) __bf16 v16bf;
typedef __attribute__((ext_vector_type(8)))  float  v8f;

static __device__ __forceinline__ unsigned int f32_bf16(float x) {
    unsigned int u = __float_as_uint(x);
    u += 0x7FFFu + ((u >> 16) & 1u);          // round-to-nearest-even
    return (u >> 16) & 0xFFFFu;
}

// Branchless tanh. Prefer the native gfx1250 v_tanh_f32 trans op; otherwise
// tanh(x) = 1 - 2*rcp(exp(2x)+1) on the hardware v_exp_f32/v_rcp_f32 path.
// Saturates correctly: exp->inf => rcp->0 => +1;  exp->0 => 1-2 = -1.
static __device__ __forceinline__ float fast_tanh(float x) {
#if __has_builtin(__builtin_amdgcn_tanhf)
    return __builtin_amdgcn_tanhf(x);
#elif __has_builtin(__builtin_amdgcn_tanh_f32)
    return __builtin_amdgcn_tanh_f32(x);
#else
    float e = __expf(2.0f * x);                       // v_exp_f32
    return 1.0f - 2.0f * __builtin_amdgcn_rcpf(e + 1.0f);  // v_rcp_f32
#endif
}

// ---------------------------------------------------------------------------
// Kernel 1: hterm[n][d] = b1[d] + sum_j h[n][j] * W1[d][CONTEXT + j]
// grid 256 (64 n x 4 quarter-rows), block 256
// ---------------------------------------------------------------------------
__global__ __launch_bounds__(256) void k_hterm(const float* __restrict__ h,
                                               const float* __restrict__ W1,
                                               const float* __restrict__ b1,
                                               float* __restrict__ hterm) {
    __shared__ float sh[HIDDEN];
    int n     = blockIdx.x >> 2;
    int dbase = (blockIdx.x & 3) * 256;
    for (int i = threadIdx.x; i < HIDDEN; i += 256) sh[i] = h[n * HIDDEN + i];
    __syncthreads();
    int d = dbase + threadIdx.x;
    const float4* wr = (const float4*)(W1 + (size_t)d * FDIM + CONTEXT);
    float acc = b1[d];
#pragma unroll 4
    for (int j = 0; j < HIDDEN / 4; ++j) {
        float4 w = wr[j];
        const float* hs = &sh[j * 4];
        acc += w.x * hs[0] + w.y * hs[1] + w.z * hs[2] + w.w * hs[3];
    }
    hterm[n * HIDDEN + d] = acc;
}

// ---------------------------------------------------------------------------
// Kernel 2: pack W1[:, :CONTEXT] -> bf16, pre-swizzled into per-lane WMMA
// B-fragment order:  w1pack dword index = ((dt*32 + ks)*32 + lane)*8 + dw
// Lane layout mirrors the 16-bit A layout transposed: column n = dt*16+(lane&15),
// lanes 0-15 hold K = kb+{0..7,16..23}, lanes 16-31 hold K = kb+{8..15,24..31}.
// grid 256, block 256 (one thread = one lane-fragment of 8 dwords)
// ---------------------------------------------------------------------------
__global__ __launch_bounds__(256) void k_pack(const float* __restrict__ W1,
                                              unsigned int* __restrict__ w1pack) {
    int tg   = blockIdx.x * 256 + threadIdx.x;   // 0 .. 65535
    int lane = tg & 31;
    int d    = ((tg >> 10) * 16) + (lane & 15);  // dt*16 + col
    int kb   = (((tg >> 5) & 31) * 32) + 8 * (lane >> 4);
    const float* wr = W1 + (size_t)d * FDIM;     // first CONTEXT entries = v-weights
    unsigned int o[8];
#pragma unroll
    for (int dw = 0; dw < 8; ++dw) {
        int k0 = kb + ((dw < 4) ? (2 * dw) : (16 + 2 * (dw - 4)));
        o[dw] = f32_bf16(wr[k0]) | (f32_bf16(wr[k0 + 1]) << 16);
    }
    unsigned int* dst = w1pack + (size_t)tg * 8;
    *(uint4*)(dst)     = make_uint4(o[0], o[1], o[2], o[3]);
    *(uint4*)(dst + 4) = make_uint4(o[4], o[5], o[6], o[7]);
}

// ---------------------------------------------------------------------------
// Kernel 3: scores[n][l] = b2 + sum_d W2[d]*tanh( vf[n,l,:].W1v[d,:] + hterm[n,d] )
// Block = 256 threads = 8 wave32, handles 128 L rows of one batch n.
// vf rows staged once to LDS as bf16 (padded stride, bank-conflict-free b128
// reads); B fragments streamed from pre-packed global (L2-resident, shared
// across waves via WGP$ since all waves walk the same d-tile sequence).
// grid 512 = 64 n x 8 row-groups, dynamic LDS = ROWS*AW*4 bytes (~258 KB)
// ---------------------------------------------------------------------------
__global__ __launch_bounds__(256) void k_score(const float* __restrict__ v,
                                               const unsigned int* __restrict__ w1pack,
                                               const float* __restrict__ hterm,
                                               const float* __restrict__ W2,
                                               const float* __restrict__ b2,
                                               float* __restrict__ scores) {
    extern __shared__ unsigned int shA[];        // ROWS * AW dwords
    int n   = blockIdx.x >> 3;
    int Lb  = (blockIdx.x & 7) * ROWS;
    int tid = threadIdx.x;

    // Stage A: 128 rows x 1024 f32 -> bf16 in LDS (coalesced float4 loads)
    {
        const float4* src = (const float4*)(v + ((size_t)n * LSP + Lb) * CONTEXT);
        for (int idx = tid; idx < ROWS * 256; idx += 256) {
            int r = idx >> 8, c4 = idx & 255;
            float4 x = src[r * 256 + c4];
            unsigned int u0 = f32_bf16(x.x) | (f32_bf16(x.y) << 16);
            unsigned int u1 = f32_bf16(x.z) | (f32_bf16(x.w) << 16);
            unsigned int* dst = &shA[r * AW + c4 * 2];
            dst[0] = u0; dst[1] = u1;
        }
    }
    __syncthreads();

    int wave = tid >> 5, lane = tid & 31;
    int lrow = lane & 15;                        // A row-in-tile / C column
    int kh   = lane >> 4;                        // which K half-stripe
    float p[8];
#pragma unroll
    for (int r = 0; r < 8; ++r) p[r] = 0.f;

    const unsigned int* arow = &shA[(wave * 16 + lrow) * AW];

    for (int dt = 0; dt < 64; ++dt) {
        int dcol  = dt * 16 + lrow;              // C-layout: N = lane&15 for all 8 VGPRs
        float ht  = hterm[n * HIDDEN + dcol];
        float w2d = W2[dcol];
        v8f c;
#pragma unroll
        for (int r = 0; r < 8; ++r) c[r] = ht;   // seed accumulator with hterm + b1

        const uint4* bsrc = (const uint4*)(w1pack + (size_t)dt * 8192);
#pragma unroll 8
        for (int ks = 0; ks < 32; ++ks) {
            union { uint4 q[2]; v16bf vv; } A, B;
            int koff = ks * 16 + kh * 4;                       // dword offset in row
            A.q[0] = *(const uint4*)(arow + koff);             // K = kb+8*kh+{0..7}
            A.q[1] = *(const uint4*)(arow + koff + 8);         // K = +16
            const uint4* bp = bsrc + ((size_t)(ks * 32 + lane)) * 2;
            B.q[0] = bp[0];
            B.q[1] = bp[1];
            c = __builtin_amdgcn_wmma_f32_16x16x32_bf16(false, A.vv, false, B.vv,
                                                        (short)0, c, false, false);
        }
#pragma unroll
        for (int r = 0; r < 8; ++r) p[r] += w2d * fast_tanh(c[r]);
    }

    // Reduce partial scores across the 16 lanes of each half (rows 0-7 / 8-15)
#pragma unroll
    for (int r = 0; r < 8; ++r) {
        float x = p[r];
        x += __shfl_xor(x, 1, 32);
        x += __shfl_xor(x, 2, 32);
        x += __shfl_xor(x, 4, 32);
        x += __shfl_xor(x, 8, 32);
        p[r] = x;
    }
    if (lrow == 0) {
        float bb = b2[0];
        int m0 = kh * 8;
#pragma unroll
        for (int r = 0; r < 8; ++r)
            scores[(size_t)n * LSP + Lb + wave * 16 + m0 + r] = p[r] + bb;
    }
}

// ---------------------------------------------------------------------------
// Kernel 4: per-n softmax over L, then out[n][c] = sum_l attn[l]*vf[n,l,c]
// grid 64 (one block per n), block 256; pure bandwidth pass (float4 coalesced)
// ---------------------------------------------------------------------------
__global__ __launch_bounds__(256) void k_softmax_out(const float* __restrict__ scores,
                                                     const float* __restrict__ v,
                                                     float* __restrict__ out) {
    __shared__ float sa[LSP];
    __shared__ float red[256];
    int n = blockIdx.x, t = threadIdx.x;

    float m = -INFINITY;
    for (int i = t; i < LSP; i += 256) { float x = scores[n * LSP + i]; sa[i] = x; m = fmaxf(m, x); }
    red[t] = m; __syncthreads();
    for (int s = 128; s > 0; s >>= 1) { if (t < s) red[t] = fmaxf(red[t], red[t + s]); __syncthreads(); }
    float mx = red[0]; __syncthreads();

    float sum = 0.f;
    for (int i = t; i < LSP; i += 256) { float e = __expf(sa[i] - mx); sa[i] = e; sum += e; }
    red[t] = sum; __syncthreads();
    for (int s = 128; s > 0; s >>= 1) { if (t < s) red[t] += red[t + s]; __syncthreads(); }
    float inv = 1.0f / red[0]; __syncthreads();

    float4 acc = make_float4(0.f, 0.f, 0.f, 0.f);
    const float4* vp = (const float4*)(v + (size_t)n * LSP * CONTEXT) + t;  // row stride 256
    for (int l = 0; l < LSP; ++l) {
        float w = sa[l];
        float4 x = vp[(size_t)l * 256];
        acc.x += w * x.x; acc.y += w * x.y; acc.z += w * x.z; acc.w += w * x.w;
    }
    float4* op = (float4*)(out + (size_t)n * CONTEXT) + t;
    *op = make_float4(acc.x * inv, acc.y * inv, acc.z * inv, acc.w * inv);
}

// ---------------------------------------------------------------------------
extern "C" void kernel_launch(void* const* d_in, const int* in_sizes, int n_in,
                              void* d_out, int out_size, void* d_ws, size_t ws_size,
                              hipStream_t stream) {
    const float* h  = (const float*)d_in[0];
    const float* v  = (const float*)d_in[1];
    const float* W1 = (const float*)d_in[2];
    const float* b1 = (const float*)d_in[3];
    const float* W2 = (const float*)d_in[4];
    const float* b2 = (const float*)d_in[5];
    float* out = (float*)d_out;

    // Workspace layout: hterm 256KB | scores 256KB | w1pack (bf16, swizzled) 2MB
    char* ws = (char*)d_ws;
    float*        hterm  = (float*)(ws);
    float*        scores = (float*)(ws + (256 << 10));
    unsigned int* w1pack = (unsigned int*)(ws + (512 << 10));

    k_hterm<<<256, 256, 0, stream>>>(h, W1, b1, hterm);
    k_pack <<<256, 256, 0, stream>>>(W1, w1pack);
    size_t ldsA = (size_t)ROWS * AW * sizeof(unsigned int);   // ~258 KB of 320 KB/WGP
    k_score<<<512, 256, ldsA, stream>>>(v, w1pack, hterm, W2, b2, scores);
    k_softmax_out<<<64, 256, 0, stream>>>(scores, v, out);
}
